// CausalSelfAttention_13838384627794
// MI455X (gfx1250) — matrix operand order repaired and checked by reading, not compiled, not run on hardware
//
#include <hip/hip_runtime.h>

// ---------------------------------------------------------------------------
// CDNA5 (gfx1250) causal self-attention, FP16 WMMA path, flash-attention style.
//   T = 4096, D = 1024, H = 16, hd = 64
// GEMMs: 32x64 per-wave tiles; k-loop unrolled x2 with ping-pong slab register
//        sets (no register rotation movs, no WMMA->VALU hazard nops).
// Attention: 4-wave block = 64 query rows sharing LDS-staged K/V (double
//        buffered, 1 barrier/iter) -> 4x less L2 K/V traffic.
// ---------------------------------------------------------------------------

#define T_SEQ   4096
#define DMODEL  1024
#define NHEAD   16
#define HDIM    64

typedef _Float16 h16;
typedef __attribute__((ext_vector_type(16))) _Float16 v16h;
typedef __attribute__((ext_vector_type(8)))  _Float16 v8h;
typedef __attribute__((ext_vector_type(8)))  float    v8f;

__device__ __forceinline__ v8f zero8() {
  v8f z;
#pragma unroll
  for (int i = 0; i < 8; ++i) z[i] = 0.0f;
  return z;
}

__device__ __forceinline__ v8f wmma_f16(v16h a, v16h b, v8f c) {
  return __builtin_amdgcn_wmma_f32_16x16x32_f16(
      false, a, false, b, (short)0, c, false, false);
}

// A-tile (16x32, MxK), row-major source. A layout (matches B):
// lane L holds row M=L&15; lanes 0-15: K 0-15, lanes 16-31: K 16-31.
__device__ __forceinline__ v16h load_tile_a(const h16* src, int ld, int lane) {
  int row = lane & 15;
  int kb  = (lane >> 4) << 4;   // 0 or 16
  const h16* p = src + (long)row * ld;
  return *(const v16h*)(p + kb);
}

// B-tile (32x16, KxN): source row n supplies B column n; K contiguous per
// half-wave (lanes 0-15 -> K 0..15, lanes 16-31 -> K 16..31): one 32B load.
__device__ __forceinline__ v16h load_tile_b(const h16* src, int ld, int lane) {
  int n  = lane & 15;
  int kb = (lane >> 4) << 4;    // 0 or 16
  return *(const v16h*)(src + (long)n * ld + kb);
}

// half-wave row reductions (masks 1,2,4,8 stay within each 16-lane half)
__device__ __forceinline__ float rowmax16(float v) {
  v = fmaxf(v, __shfl_xor(v, 1, 32));
  v = fmaxf(v, __shfl_xor(v, 2, 32));
  v = fmaxf(v, __shfl_xor(v, 4, 32));
  v = fmaxf(v, __shfl_xor(v, 8, 32));
  return v;
}
__device__ __forceinline__ float rowsum16(float v) {
  v += __shfl_xor(v, 1, 32);
  v += __shfl_xor(v, 2, 32);
  v += __shfl_xor(v, 4, 32);
  v += __shfl_xor(v, 8, 32);
  return v;
}

// ---------------------------------------------------------------------------
// GEMM slab helpers: one slab = 32 k-values of a 32(M)x64(N) wave tile.
// ---------------------------------------------------------------------------
struct Slab { v16h A0, A1, B0, B1, B2, B3; };

__device__ __forceinline__ void load_slab(Slab& s, const h16* a0, const h16* a1,
                                          const h16* brow, int k, int lane) {
  s.A0 = load_tile_a(a0 + k, DMODEL, lane);
  s.A1 = load_tile_a(a1 + k, DMODEL, lane);
  s.B0 = load_tile_b(brow + 0L * 16 * DMODEL + k, DMODEL, lane);
  s.B1 = load_tile_b(brow + 1L * 16 * DMODEL + k, DMODEL, lane);
  s.B2 = load_tile_b(brow + 2L * 16 * DMODEL + k, DMODEL, lane);
  s.B3 = load_tile_b(brow + 3L * 16 * DMODEL + k, DMODEL, lane);
}

__device__ __forceinline__ void compute_slab(v8f acc[8], const Slab& s) {
  acc[0] = wmma_f16(s.A0, s.B0, acc[0]);
  acc[4] = wmma_f16(s.A1, s.B0, acc[4]);
  acc[1] = wmma_f16(s.A0, s.B1, acc[1]);
  acc[5] = wmma_f16(s.A1, s.B1, acc[5]);
  acc[2] = wmma_f16(s.A0, s.B2, acc[2]);
  acc[6] = wmma_f16(s.A1, s.B2, acc[6]);
  acc[3] = wmma_f16(s.A0, s.B3, acc[3]);
  acc[7] = wmma_f16(s.A1, s.B3, acc[7]);
}

// ---------------------------------------------------------------------------
__global__ void cvt_f32_f16(const float* __restrict__ in, h16* __restrict__ out, long n) {
  long i = (long)blockIdx.x * blockDim.x + threadIdx.x;
  long stride = (long)gridDim.x * blockDim.x;
  for (; i < n; i += stride) out[i] = (h16)in[i];
}

// ---------------------------------------------------------------------------
// QKV projection: 32(M) x 64(N) tile per wave, ping-pong pipelined k loop.
// Outputs: Q, K as [T][1024] f16; V transposed as [H*64][T] f16.
// ---------------------------------------------------------------------------
__global__ __launch_bounds__(128) void qkv_gemm(
    const h16* __restrict__ xb, const h16* __restrict__ wq,
    const float* __restrict__ bias,
    h16* __restrict__ Qb, h16* __restrict__ Kb, h16* __restrict__ Vt) {
  int lane = threadIdx.x & 31;
  int w    = blockIdx.x * (blockDim.x >> 5) + (threadIdx.x >> 5);
  int tm   = w & 127;           // 128 M-tiles (4096/32)
  int tn   = w >> 7;            // 48  N-tiles (3072/64)
  int m0 = tm * 32, n0 = tn * 64;

  const h16* a0 = xb + (long)m0 * DMODEL;
  const h16* a1 = a0 + 16L * DMODEL;
  const h16* brow = wq + (long)n0 * DMODEL;

  v8f acc[8];
#pragma unroll
  for (int t = 0; t < 8; ++t) acc[t] = zero8();

  Slab s0, s1;
  load_slab(s0, a0, a1, brow, 0, lane);
  for (int k = 0; k < DMODEL - 64; k += 64) {
    load_slab(s1, a0, a1, brow, k + 32, lane);  // prefetch odd slab
    compute_slab(acc, s0);                      // compute even slab
    load_slab(s0, a0, a1, brow, k + 64, lane);  // prefetch next even slab
    compute_slab(acc, s1);                      // compute odd slab
  }
  load_slab(s1, a0, a1, brow, DMODEL - 32, lane);
  compute_slab(acc, s0);
  compute_slab(acc, s1);

  int n = lane & 15, hi = lane >> 4;
#pragma unroll
  for (int half = 0; half < 2; ++half) {
#pragma unroll
    for (int t = 0; t < 4; ++t) {
      int col = n0 + t * 16 + n;
      float bv = bias[col];
#pragma unroll
      for (int r = 0; r < 8; ++r) {
        int row = m0 + half * 16 + r + 8 * hi;
        h16 v = (h16)(acc[half * 4 + t][r] + bv);
        if (col < DMODEL)          Qb[(long)row * DMODEL + col] = v;
        else if (col < 2 * DMODEL) Kb[(long)row * DMODEL + (col - DMODEL)] = v;
        else                       Vt[(long)(col - 2 * DMODEL) * T_SEQ + row] = v;
      }
    }
  }
}

// ---------------------------------------------------------------------------
// Flash attention: block = 4 waves x 16 query rows (64 consecutive queries),
// sharing double-buffered K/V tiles in LDS. Pipeline per key block j:
//   issue global loads for j+1 -> compute S/softmax/PV from LDS buf[j&1]
//   -> ds-store j+1 into buf[(j+1)&1] -> barrier.
// ---------------------------------------------------------------------------
__global__ __launch_bounds__(128) void attn_kernel(
    const h16* __restrict__ Qb, const h16* __restrict__ Kb,
    const h16* __restrict__ Vt, h16* __restrict__ Yatt) {
  __shared__ h16 Ksh[2][32][64];      // [buf][key][dim]   4KB each
  __shared__ h16 Vsh[2][64][32];      // [buf][dim][key]   4KB each
  __shared__ h16 Psh[4][16 * 32];     // per-wave P staging 1KB each

  int tid  = threadIdx.x;
  int lane = tid & 31;
  int wave = tid >> 5;
  int head = blockIdx.y;
  int qb0  = blockIdx.x * 64;         // block's query range
  int q0   = qb0 + wave * 16;         // this wave's 16 queries
  int n = lane & 15, hi = lane >> 4;

  // per-thread staging coords: K 32 rows x 128B, V 64 rows x 64B; 32B/thread
  int kr = tid >> 2, kc = (tid & 3) * 16;
  int vd = tid >> 1, vk = (tid & 1) * 16;
  const h16* kg = Kb + (long)head * HDIM;           // + (k0+kr)*DMODEL + kc
  const h16* vg = Vt + (long)head * HDIM * T_SEQ;   // + vd*T_SEQ + k0 + vk

  int nkb = (qb0 + 64 + 31) / 32;     // key blocks covering keys <= qb0+63

  const h16* qsrc = Qb + (long)q0 * DMODEL + head * HDIM;
  v16h Aq0 = load_tile_a(qsrc,      DMODEL, lane);
  v16h Aq1 = load_tile_a(qsrc + 32, DMODEL, lane);

  // prologue: stage key block 0
  {
    v16h kv = *(const v16h*)(kg + (long)kr * DMODEL + kc);
    v16h vv = *(const v16h*)(vg + (long)vd * T_SEQ + vk);
    *(v16h*)&Ksh[0][kr][kc] = kv;
    *(v16h*)&Vsh[0][vd][vk] = vv;
  }
  __syncthreads();

  v8f O[4];
#pragma unroll
  for (int t = 0; t < 4; ++t) O[t] = zero8();
  float m[8], l[8];
#pragma unroll
  for (int r = 0; r < 8; ++r) { m[r] = -__builtin_inff(); l[r] = 0.0f; }

  const float scale = 0.125f;         // 1/sqrt(64)

  for (int j = 0; j < nkb; ++j) {
    int k0  = j * 32;
    int buf = j & 1;
    bool pf = (j + 1 < nkb);

    // issue next block's global loads first: latency hidden by compute below
    v16h knext{}, vnext{};
    if (pf) {
      knext = *(const v16h*)(kg + (long)(k0 + 32 + kr) * DMODEL + kc);
      vnext = *(const v16h*)(vg + (long)vd * T_SEQ + (k0 + 32) + vk);
    }

    // S = Q.K^T from LDS (keys n / 16+n, dims split 0-31 / 32-63)
    const h16* kt = &Ksh[buf][0][0];
    v8f S0 = wmma_f16(Aq0, load_tile_b(kt,             HDIM, lane), zero8());
    S0     = wmma_f16(Aq1, load_tile_b(kt + 32,        HDIM, lane), S0);
    v8f S1 = wmma_f16(Aq0, load_tile_b(kt + 16 * HDIM,      HDIM, lane), zero8());
    S1     = wmma_f16(Aq1, load_tile_b(kt + 16 * HDIM + 32, HDIM, lane), S1);

    // scale + causal mask + online softmax (row r -> query q0 + r + 8*hi)
#pragma unroll
    for (int r = 0; r < 8; ++r) {
      int qrow = q0 + r + 8 * hi;
      float s0 = S0[r] * scale; if (k0 + n      > qrow) s0 = -__builtin_inff();
      float s1 = S1[r] * scale; if (k0 + 16 + n > qrow) s1 = -__builtin_inff();
      float mn = fmaxf(m[r], rowmax16(fmaxf(s0, s1)));
      float a  = __expf(m[r] - mn);
      float p0 = __expf(s0 - mn);
      float p1 = __expf(s1 - mn);
      S0[r] = p0; S1[r] = p1;
      l[r] = a * l[r] + rowsum16(p0 + p1);
      m[r] = mn;
#pragma unroll
      for (int t = 0; t < 4; ++t) O[t][r] *= a;
    }

    // C-layout -> A-layout for P (wave-private LDS roundtrip)
    h16* pl = Psh[wave];
#pragma unroll
    for (int r = 0; r < 8; ++r) {
      int row = r + 8 * hi;
      pl[row * 32 + n]      = (h16)S0[r];
      pl[row * 32 + 16 + n] = (h16)S1[r];
    }
    v16h Ap = load_tile_a(pl, 32, lane);

    // O += P(16x32) . V(32x64) from LDS (Vsh is [dim][key])
    const h16* vt = &Vsh[buf][0][0];
#pragma unroll
    for (int t = 0; t < 4; ++t) {
      v16h Bv = load_tile_b(vt + t * 16 * 32, 32, lane);
      O[t] = wmma_f16(Ap, Bv, O[t]);
    }

    // land next block's staging, then sync (also fences buf reuse at j+2)
    if (pf) {
      *(v16h*)&Ksh[buf ^ 1][kr][kc] = knext;
      *(v16h*)&Vsh[buf ^ 1][vd][vk] = vnext;
    }
    __syncthreads();
  }

  float linv[8];
#pragma unroll
  for (int r = 0; r < 8; ++r) linv[r] = 1.0f / l[r];
#pragma unroll
  for (int t = 0; t < 4; ++t) {
#pragma unroll
    for (int r = 0; r < 8; ++r) {
      int qrow = q0 + r + 8 * hi;
      Yatt[(long)qrow * DMODEL + head * HDIM + t * 16 + n] = (h16)(O[t][r] * linv[r]);
    }
  }
}

// ---------------------------------------------------------------------------
// Output projection: 32x64 tile per wave, ping-pong pipelined, fp32 out.
// ---------------------------------------------------------------------------
__global__ __launch_bounds__(128) void proj_gemm(
    const h16* __restrict__ ya, const h16* __restrict__ wp,
    const float* __restrict__ bias, float* __restrict__ out) {
  int lane = threadIdx.x & 31;
  int w    = blockIdx.x * (blockDim.x >> 5) + (threadIdx.x >> 5);
  int tm   = w & 127;           // 128 M-tiles
  int tn   = w >> 7;            // 16 N-tiles
  int m0 = tm * 32, n0 = tn * 64;

  const h16* a0 = ya + (long)m0 * DMODEL;
  const h16* a1 = a0 + 16L * DMODEL;
  const h16* brow = wp + (long)n0 * DMODEL;

  v8f acc[8];
#pragma unroll
  for (int t = 0; t < 8; ++t) acc[t] = zero8();

  Slab s0, s1;
  load_slab(s0, a0, a1, brow, 0, lane);
  for (int k = 0; k < DMODEL - 64; k += 64) {
    load_slab(s1, a0, a1, brow, k + 32, lane);
    compute_slab(acc, s0);
    load_slab(s0, a0, a1, brow, k + 64, lane);
    compute_slab(acc, s1);
  }
  load_slab(s1, a0, a1, brow, DMODEL - 32, lane);
  compute_slab(acc, s0);
  compute_slab(acc, s1);

  int n = lane & 15, hi = lane >> 4;
#pragma unroll
  for (int half = 0; half < 2; ++half) {
#pragma unroll
    for (int t = 0; t < 4; ++t) {
      int col = n0 + t * 16 + n;
      float bv = bias[col];
#pragma unroll
      for (int r = 0; r < 8; ++r) {
        int row = m0 + half * 16 + r + 8 * hi;
        out[(long)row * DMODEL + col] = acc[half * 4 + t][r] + bv;
      }
    }
  }
}

// ---------------------------------------------------------------------------
extern "C" void kernel_launch(void* const* d_in, const int* in_sizes, int n_in,
                              void* d_out, int out_size, void* d_ws, size_t ws_size,
                              hipStream_t stream) {
  const float* x      = (const float*)d_in[0];   // (4096, 1024)
  const float* W_qkv  = (const float*)d_in[1];   // (3072, 1024)
  const float* b_qkv  = (const float*)d_in[2];   // (3072,)
  const float* W_proj = (const float*)d_in[3];   // (1024, 1024)
  const float* b_proj = (const float*)d_in[4];   // (1024,)
  float* out = (float*)d_out;

  char* ws = (char*)d_ws;
  h16* xb  = (h16*)ws;                   ws += (long)T_SEQ * DMODEL * 2;       // 8 MiB
  h16* wqb = (h16*)ws;                   ws += (long)3 * DMODEL * DMODEL * 2;  // 6 MiB
  h16* wpb = (h16*)ws;                   ws += (long)DMODEL * DMODEL * 2;      // 2 MiB
  h16* Qb  = (h16*)ws;                   ws += (long)T_SEQ * DMODEL * 2;       // 8 MiB
  h16* Kb  = (h16*)ws;                   ws += (long)T_SEQ * DMODEL * 2;       // 8 MiB
  h16* Vt  = (h16*)ws;                   ws += (long)T_SEQ * DMODEL * 2;       // 8 MiB
  h16* Ya  = (h16*)ws;                                                        // 8 MiB

  cvt_f32_f16<<<4096, 256, 0, stream>>>(x,      xb,  (long)T_SEQ * DMODEL);
  cvt_f32_f16<<<4096, 256, 0, stream>>>(W_qkv,  wqb, (long)3 * DMODEL * DMODEL);
  cvt_f32_f16<<<2048, 256, 0, stream>>>(W_proj, wpb, (long)DMODEL * DMODEL);

  // QKV: 128 x 48 wave-tiles, 4 waves / block
  qkv_gemm<<<(128 * 48) / 4, 128, 0, stream>>>(xb, wqb, b_qkv, Qb, Kb, Vt);

  // attention: 64 query-blocks (64 rows each) x 16 heads
  attn_kernel<<<dim3(64, NHEAD), 128, 0, stream>>>(Qb, Kb, Vt, Ya);

  // projection: 128 x 16 wave-tiles
  proj_gemm<<<(128 * 16) / 4, 128, 0, stream>>>(Ya, wpb, b_proj, out);
}
